// SelfAttention_88021059764447
// MI455X (gfx1250) — matrix operand order repaired and checked by reading, not compiled, not run on hardware
//
#include <hip/hip_runtime.h>

// ---------------------------------------------------------------------------
// Types matching gfx1250 WMMA / async builtin signatures
// ---------------------------------------------------------------------------
typedef __bf16 v16bf __attribute__((ext_vector_type(16)));
typedef __bf16 v8bf  __attribute__((ext_vector_type(8)));
typedef __bf16 v4bf  __attribute__((ext_vector_type(4)));
typedef float  v8f   __attribute__((ext_vector_type(8)));
typedef int    v4i   __attribute__((ext_vector_type(4)));

#define WMMA_BF16(A, B, C) \
    __builtin_amdgcn_wmma_f32_16x16x32_bf16(false, (A), false, (B), (short)0, (C), false, false)

static constexpr int BB = 4;
static constexpr int TT = 2048;
static constexpr int EE = 1024;
static constexpr int HH = 16;
static constexpr int DH = 64;
static constexpr float SCALE = 0.125f; // 1/sqrt(64)

// ---------------------------------------------------------------------------
// CDNA5 async global->LDS copy path (ASYNCcnt), with safe fallback
// ---------------------------------------------------------------------------
#if defined(__AMDGCN__) && __has_builtin(__builtin_amdgcn_global_load_async_to_lds_b128)
#define USE_ASYNC 1
#else
#define USE_ASYNC 0
#endif

__device__ __forceinline__ void copy16(const __bf16* g, __bf16* l) {
#if USE_ASYNC
    __builtin_amdgcn_global_load_async_to_lds_b128(
        (__attribute__((address_space(1))) v4i*)g,
        (__attribute__((address_space(3))) v4i*)l, 0, 0);
#else
    *(v8bf*)l = *(const v8bf*)g;
#endif
}

#if USE_ASYNC && __has_builtin(__builtin_amdgcn_s_wait_asynccnt)
#define ASYNC_WAIT(n) __builtin_amdgcn_s_wait_asynccnt(n)
#else
#define ASYNC_WAIT(n) ((void)0)
#endif

__device__ __forceinline__ v16bf cat8(v8bf a, v8bf b) {
    return __builtin_shufflevector(a, b, 0,1,2,3,4,5,6,7,8,9,10,11,12,13,14,15);
}

// ---------------------------------------------------------------------------
// Pre-pass: elementwise fp32 -> bf16
// ---------------------------------------------------------------------------
__global__ __launch_bounds__(256)
void convert_f32_bf16(const float* __restrict__ src, __bf16* __restrict__ dst, int n) {
    int i = (blockIdx.x * 256 + threadIdx.x) * 4;
    if (i >= n) return;
    const float4 f = *(const float4*)(src + i);
    v4bf o;
    o[0] = (__bf16)f.x; o[1] = (__bf16)f.y; o[2] = (__bf16)f.z; o[3] = (__bf16)f.w;
    *(v4bf*)(dst + i) = o;
}

// ---------------------------------------------------------------------------
// Pre-pass: W [K][N] fp32 -> Wt [N][K] bf16 (so GEMM B-tiles are k-contiguous)
// ---------------------------------------------------------------------------
template<int K, int N>
__global__ __launch_bounds__(256)
void transpose_to_bf16(const float* __restrict__ src, __bf16* __restrict__ dst) {
    __shared__ float tile[32][33];
    const int bn = blockIdx.x * 32, bk = blockIdx.y * 32;
    const int tx = threadIdx.x & 31, ty = threadIdx.x >> 5;   // 32 x 8
    #pragma unroll
    for (int i = 0; i < 32; i += 8)
        tile[ty + i][tx] = src[(size_t)(bk + ty + i) * N + bn + tx];
    __syncthreads();
    #pragma unroll
    for (int i = 0; i < 32; i += 8)
        dst[(size_t)(bn + ty + i) * K + bk + tx] = (__bf16)tile[tx][ty + i];
}

// ---------------------------------------------------------------------------
// GEMM: C[M,N] = A[M,K](bf16) * Bt[N,K]^T(bf16) + bias[N]
// Block: 256 threads (8 waves). Tile 128x128, BK = 32, double-buffered LDS.
// Waves 4x2; each wave computes a 32x64 tile (2x4 WMMA accumulators).
// Software pipeline: async-stage tile k+1 while computing tile k
// (4 async b128 loads / thread / tile, in-order completion -> wait <= 4).
// ---------------------------------------------------------------------------
template<int K, int N, bool OUT_BF16>
__global__ __launch_bounds__(256)
void gemm_bf16(const __bf16* __restrict__ A, const __bf16* __restrict__ Bt,
               const float* __restrict__ bias, void* __restrict__ Cv) {
    __shared__ __align__(16) __bf16 sA[2][128 * 32];   // [m][k]
    __shared__ __align__(16) __bf16 sB[2][128 * 32];   // [n][k]

    const int tid  = threadIdx.x;
    const int lane = tid & 31;
    const int wave = tid >> 5;
    const int wm   = wave >> 1;     // 0..3 -> row offset wm*32
    const int wn   = wave & 1;      // 0..1 -> col offset wn*64
    const int hi   = lane >> 4;
    const int ln   = lane & 15;

    const int bm = blockIdx.y * 128;
    const int bn = blockIdx.x * 128;

    // per-thread staging coordinates (2 chunks for A, 2 for B)
    const int cr = tid >> 2, cp = tid & 3;

    auto stage = [&](int buf, int k0) {
        #pragma unroll
        for (int i = 0; i < 2; ++i) {
            int r = cr + i * 64;
            copy16(A + (size_t)(bm + r) * K + k0 + cp * 8, &sA[buf][r * 32 + cp * 8]);
        }
        #pragma unroll
        for (int i = 0; i < 2; ++i) {
            int r = cr + i * 64;
            copy16(Bt + (size_t)(bn + r) * K + k0 + cp * 8, &sB[buf][r * 32 + cp * 8]);
        }
    };

    v8f acc[2][4] = {};

    constexpr int NK = K / 32;
    stage(0, 0);
    for (int kt = 0; kt < NK; ++kt) {
        const int b = kt & 1;
        if (kt + 1 < NK) {
            stage(b ^ 1, (kt + 1) * 32);   // overlap next tile's DMA with compute
            ASYNC_WAIT(4);                 // tile kt's 4 loads complete (in-order)
        } else {
            ASYNC_WAIT(0);
        }
        __syncthreads();

        v16bf afrag[2], bfrag[4];
        #pragma unroll
        for (int i = 0; i < 2; ++i) {
            const __bf16* p = &sA[b][(wm * 32 + i * 16 + ln) * 32 + hi * 8];
            afrag[i] = cat8(*(const v8bf*)p, *(const v8bf*)(p + 16));
        }
        #pragma unroll
        for (int j = 0; j < 4; ++j) {
            const __bf16* p = &sB[b][(wn * 64 + j * 16 + ln) * 32 + hi * 8];
            bfrag[j] = cat8(*(const v8bf*)p, *(const v8bf*)(p + 16));
        }
        #pragma unroll
        for (int i = 0; i < 2; ++i)
            #pragma unroll
            for (int j = 0; j < 4; ++j)
                acc[i][j] = WMMA_BF16(afrag[i], bfrag[j], acc[i][j]);
        __syncthreads();   // release buffer b before it is re-staged
    }

    // epilogue: C-layout row = r + 8*hi, col = ln; N is compile-time constant
    #pragma unroll
    for (int i = 0; i < 2; ++i) {
        const int row0 = bm + wm * 32 + i * 16 + hi * 8;
        #pragma unroll
        for (int j = 0; j < 4; ++j) {
            const int col = bn + wn * 64 + j * 16 + ln;
            const float bv = bias[col];
            if constexpr (OUT_BF16) {
                __bf16* C = (__bf16*)Cv;
                #pragma unroll
                for (int r = 0; r < 8; ++r)
                    C[(size_t)(row0 + r) * N + col] = (__bf16)(acc[i][j][r] + bv);
            } else {
                float* C = (float*)Cv;
                #pragma unroll
                for (int r = 0; r < 8; ++r)
                    C[(size_t)(row0 + r) * N + col] = acc[i][j][r] + bv;
            }
        }
    }
}

// ---------------------------------------------------------------------------
// Fused causal flash attention over bf16 qkv [B*T, 3E]; emits bf16 y [B*T, E].
// Grid (T/64, H, B); 128 threads = 4 waves, wave w owns rows q0+16w..+15.
// Double-buffered K/V staging: K via async b128 copies (ASYNCcnt pipeline),
// V via global loads into registers at iteration top (latency overlapped with
// compute) + transposed ds stores at iteration bottom. The P round-trip uses
// wave-private LDS (per-wave in-order DS => no block barrier needed).
// ---------------------------------------------------------------------------
__global__ __launch_bounds__(128)
void attn_fwd(const __bf16* __restrict__ qkv, __bf16* __restrict__ y) {
    const int qt = blockIdx.x, h = blockIdx.y, b = blockIdx.z;
    const int q0 = qt * 64;
    const int tid = threadIdx.x, lane = tid & 31, wave = tid >> 5;
    const int hi = lane >> 4, ln = lane & 15;

    __shared__ __align__(16) __bf16 sK[2][32 * 64];   // [key][d]
    __shared__ __align__(16) __bf16 sVT[2][64 * 32];  // [d][key]
    __shared__ __align__(16) __bf16 sP[4][16 * 32];   // per-wave P [row][key]

    constexpr int RS = 3 * EE;                         // qkv row stride (elems)
    const __bf16* qBase = qkv + (size_t)b * TT * RS + (size_t)h * DH;
    const __bf16* kBase = qBase + EE;
    const __bf16* vBase = qBase + 2 * EE;

    // Q fragments: rows q0 + 16*wave + ln (bf16, direct b128 loads)
    v16bf qf[2];
    {
        const __bf16* qp = qBase + (size_t)(q0 + wave * 16 + ln) * RS;
        #pragma unroll
        for (int c = 0; c < 2; ++c) {
            const __bf16* p = qp + c * 32 + hi * 8;
            qf[c] = cat8(*(const v8bf*)p, *(const v8bf*)(p + 16));
        }
    }

    // staging helpers (2 x 16B chunks per thread for K and for V)
    v8bf vr[2];
    auto stageK = [&](int buf, int k0) {
        #pragma unroll
        for (int i = 0; i < 2; ++i) {
            int ch = tid + i * 128;
            int key = ch >> 3, p = ch & 7;
            copy16(kBase + (size_t)(k0 + key) * RS + p * 8, &sK[buf][key * 64 + p * 8]);
        }
    };
    auto loadV = [&](int k0) {
        #pragma unroll
        for (int i = 0; i < 2; ++i) {
            int ch = tid + i * 128;
            int key = ch >> 3, dp = (ch & 7) * 8;
            vr[i] = *(const v8bf*)(vBase + (size_t)(k0 + key) * RS + dp);
        }
    };
    auto storeV = [&](int buf) {
        #pragma unroll
        for (int i = 0; i < 2; ++i) {
            int ch = tid + i * 128;
            int key = ch >> 3, dp = (ch & 7) * 8;
            #pragma unroll
            for (int u = 0; u < 8; ++u)
                sVT[buf][(dp + u) * 32 + key] = vr[i][u];
        }
    };

    float m_run[8], l_run[8];
    v8f o[4] = {};
    #pragma unroll
    for (int r = 0; r < 8; ++r) { m_run[r] = -1e30f; l_run[r] = 0.f; }

    const int qmax_wave = q0 + wave * 16 + 15;
    const int ktiles = (q0 + 64) / 32;

    // prologue: stage tile 0 into buffer 0
    stageK(0, 0);
    loadV(0);
    storeV(0);

    for (int kt = 0; kt < ktiles; ++kt) {
        const int k0 = kt * 32;
        const int bf = kt & 1;
        if (kt + 1 < ktiles) {
            stageK(bf ^ 1, k0 + 32);   // async K for next tile
            loadV(k0 + 32);            // V global loads in flight during compute
            ASYNC_WAIT(2);             // tile kt's K loads complete (in-order)
        } else {
            ASYNC_WAIT(0);
        }
        __syncthreads();               // sK[bf]/sVT[bf] visible to all waves

        const bool active = (k0 <= qmax_wave);   // wave-uniform
        if (active) {
            // S = Q * K^T : two 16-key N-tiles, two 32-d K-steps each
            v8f s[2] = {};
            #pragma unroll
            for (int nt = 0; nt < 2; ++nt)
                #pragma unroll
                for (int c = 0; c < 2; ++c) {
                    const __bf16* p = &sK[bf][(nt * 16 + ln) * 64 + c * 32 + hi * 8];
                    v16bf kf = cat8(*(const v8bf*)p, *(const v8bf*)(p + 16));
                    s[nt] = WMMA_BF16(qf[c], kf, s[nt]);
                }

            // causal mask + online softmax; SCALE folded in here
            #pragma unroll
            for (int r = 0; r < 8; ++r) {
                const int qr = q0 + wave * 16 + r + 8 * hi;
                float v0 = (k0 + ln > qr)      ? -1e30f : s[0][r] * SCALE;
                float v1 = (k0 + 16 + ln > qr) ? -1e30f : s[1][r] * SCALE;
                float mx = fmaxf(v0, v1);
                #pragma unroll
                for (int off = 1; off < 16; off <<= 1)
                    mx = fmaxf(mx, __shfl_xor(mx, off, 32));
                const float m_new = fmaxf(m_run[r], mx);
                const float corr  = __expf(m_run[r] - m_new);
                const float e0 = __expf(v0 - m_new);
                const float e1 = __expf(v1 - m_new);
                float sum = e0 + e1;
                #pragma unroll
                for (int off = 1; off < 16; off <<= 1)
                    sum += __shfl_xor(sum, off, 32);
                l_run[r] = l_run[r] * corr + sum;
                m_run[r] = m_new;
                #pragma unroll
                for (int dt = 0; dt < 4; ++dt) o[dt][r] *= corr;
                // P: C-layout -> wave-private LDS (row = r + 8*hi)
                sP[wave][(r + 8 * hi) * 32 + ln]      = (__bf16)e0;
                sP[wave][(r + 8 * hi) * 32 + 16 + ln] = (__bf16)e1;
            }

            // per-wave DS ordering makes sP stores visible to this wave's loads
            const __bf16* pp = &sP[wave][ln * 32 + hi * 8];
            v16bf pf = cat8(*(const v8bf*)pp, *(const v8bf*)(pp + 16));
            #pragma unroll
            for (int dt = 0; dt < 4; ++dt) {
                const __bf16* vp = &sVT[bf][(dt * 16 + ln) * 32 + hi * 8];
                v16bf vf = cat8(*(const v8bf*)vp, *(const v8bf*)(vp + 16));
                o[dt] = WMMA_BF16(pf, vf, o[dt]);
            }
        }

        if (kt + 1 < ktiles) storeV(bf ^ 1);   // transpose-store next V tile
        __syncthreads();                        // release buffer bf
    }

    // finalize: y[b, q, h*64 + d] = (bf16)(o / l)
    __bf16* yp = y + ((size_t)b * TT + q0 + wave * 16) * EE + (size_t)h * DH;
    #pragma unroll
    for (int r = 0; r < 8; ++r) {
        const float inv = 1.0f / l_run[r];
        const int row = r + 8 * hi;
        #pragma unroll
        for (int dt = 0; dt < 4; ++dt)
            yp[(size_t)row * EE + dt * 16 + ln] = (__bf16)(o[dt][r] * inv);
    }
}

// ---------------------------------------------------------------------------
extern "C" void kernel_launch(void* const* d_in, const int* in_sizes, int n_in,
                              void* d_out, int out_size, void* d_ws, size_t ws_size,
                              hipStream_t stream) {
    const float* x      = (const float*)d_in[0];
    const float* W_qkv  = (const float*)d_in[1];
    const float* b_qkv  = (const float*)d_in[2];
    const float* W_proj = (const float*)d_in[3];
    const float* b_proj = (const float*)d_in[4];
    float* out = (float*)d_out;

    const int M = BB * TT;  // 8192

    // workspace carve-up (all bf16, 16B-aligned sizes)
    char* w = (char*)d_ws;
    __bf16* xb     = (__bf16*)w;  w += (size_t)M * EE * 2;          // 16 MB
    __bf16* Wqkvt  = (__bf16*)w;  w += (size_t)3 * EE * EE * 2;     //  6 MB
    __bf16* Wprojt = (__bf16*)w;  w += (size_t)EE * EE * 2;         //  2 MB
    __bf16* qkvb   = (__bf16*)w;  w += (size_t)M * 3 * EE * 2;      // 48 MB
    __bf16* yb     = (__bf16*)w;                                    // 16 MB

    // 0) precision pre-pass: x -> bf16; W -> bf16 transposed [n][k]
    {
        int n = M * EE;
        convert_f32_bf16<<<n / (4 * 256), 256, 0, stream>>>(x, xb, n);
    }
    transpose_to_bf16<EE, 3 * EE><<<dim3(3 * EE / 32, EE / 32), 256, 0, stream>>>(W_qkv, Wqkvt);
    transpose_to_bf16<EE, EE>    <<<dim3(EE / 32,     EE / 32), 256, 0, stream>>>(W_proj, Wprojt);

    // 1) qkv(bf16) = x @ W_qkv + b_qkv
    gemm_bf16<EE, 3 * EE, true><<<dim3(3 * EE / 128, M / 128), 256, 0, stream>>>(
        xb, Wqkvt, b_qkv, qkvb);

    // 2) fused causal attention -> y(bf16)
    attn_fwd<<<dim3(TT / 64, HH, BB), 128, 0, stream>>>(qkvb, yb);

    // 3) out(fp32) = y @ W_proj + b_proj
    gemm_bf16<EE, EE, false><<<dim3(EE / 128, M / 128), 256, 0, stream>>>(
        yb, Wprojt, b_proj, out);
}
